// AttentionLayer_65489661330126
// MI455X (gfx1250) — matrix operand order repaired and checked by reading, not compiled, not run on hardware
//
#include <hip/hip_runtime.h>
#include <hip/hip_bf16.h>

// ---------------------------------------------------------------------------
// MI455X (gfx1250) attention layer.
//  * all matmuls via v_wmma_f32_16x16x32_f16 (f16 in, f32 accumulate)
//  * flash-attention, scores^T orientation: softmax reduction = 1 shuffle,
//    K/V staged by async global->LDS (ASYNCcnt) with LDS double buffering
//  * f32 -> f16 conversion fused into LDS staging of the GEMMs
// ---------------------------------------------------------------------------

typedef __attribute__((ext_vector_type(16))) _Float16 v16h;
typedef __attribute__((ext_vector_type(8)))  _Float16 v8h;
typedef __attribute__((ext_vector_type(4)))  _Float16 v4h;
typedef __attribute__((ext_vector_type(8)))  float    v8f;

#define HIDDEN 1024
#define NHEAD  16
#define DHEAD  64
#define SEQ    2048
#define BATCH  2
#define MTOT   (BATCH * SEQ)

// ---- WMMA fragment helpers (layouts per CDNA5 ISA 7.12.2) ------------------
// A (16x32 f16): lane%16 = M row; g = lane/16; element e -> K = (e/8)*16 + g*8 + e%8
__device__ __forceinline__ v16h load_a_frag(const _Float16* rowp, int g) {
  v8h lo = *(const v8h*)(rowp + g * 8);
  v8h hi = *(const v8h*)(rowp + 16 + g * 8);
  return __builtin_shufflevector(lo, hi, 0, 1, 2, 3, 4, 5, 6, 7,
                                 8, 9, 10, 11, 12, 13, 14, 15);
}
// B (32x16 f16): lane%16 = N column; g = lane/16; element e -> K = (e/8)*16 + g*8 + e%8
__device__ __forceinline__ v16h load_b_frag(const _Float16* p, int stride, int g) {
  v16h r;
#pragma unroll
  for (int s = 0; s < 16; ++s) r[s] = p[((s >> 3) * 16 + g * 8 + (s & 7)) * stride];
  return r;
}
__device__ __forceinline__ v8f wmma_f16(v16h a, v16h b, v8f c) {
  return __builtin_amdgcn_wmma_f32_16x16x32_f16(false, a, false, b,
                                                (short)0, c, false, false);
}
// Async VGPR-free copy: 16B per lane, tracked by ASYNCcnt.
__device__ __forceinline__ void async_b128(unsigned ldsOff, const _Float16* gp) {
  asm volatile("global_load_async_to_lds_b128 %0, %1, off"
               :: "v"(ldsOff), "v"((unsigned long long)(uintptr_t)gp)
               : "memory");
}

// ---------------------------------------------------------------------------
// GEMM: [MTOT x 1024] x [1024 x 1024] + bias.
// Block tile 128x128, BLK_K=32, 256 threads = 8 waves (4 along M, 2 along N),
// each wave owns 2x4 = 8 accumulator fragments (32x64 output).
// ---------------------------------------------------------------------------
#define BLK_M   128
#define BLK_N   128
#define BLK_K   32
#define A_STR   40    // halves; row stride 80B (16B multiple)
#define B_STR   136   // halves; row stride 272B (16B multiple)

template <bool IN_F32, bool OUT_HEADS>
__global__ __launch_bounds__(256) void gemm_wmma(
    const void* __restrict__ Xv, const float* __restrict__ W,
    const float* __restrict__ bias, void* __restrict__ Ov, float scale) {
  __shared__ _Float16 As[BLK_M * A_STR];
  __shared__ _Float16 Bs[BLK_K * B_STR];

  const int tid   = threadIdx.x;
  const int lane  = tid & 31;
  const int wave  = tid >> 5;
  const int waveM = wave & 3;
  const int waveN = wave >> 2;
  const int g     = lane >> 4;
  const int m0    = blockIdx.x * BLK_M;
  const int n0    = blockIdx.y * BLK_N;

  const v8f vz = {0.f, 0.f, 0.f, 0.f, 0.f, 0.f, 0.f, 0.f};
  v8f acc[2][4];
#pragma unroll
  for (int wm = 0; wm < 2; ++wm)
#pragma unroll
    for (int nf = 0; nf < 4; ++nf) acc[wm][nf] = vz;

  for (int k0 = 0; k0 < HIDDEN; k0 += BLK_K) {
    // ---- stage A tile (convert f32->f16 if needed) ----
    if constexpr (IN_F32) {
      const float* X = (const float*)Xv;
#pragma unroll
      for (int i = 0; i < 4; ++i) {
        int fi = tid + i * 256;            // 1024 float4 chunks
        int row = fi >> 3, c4 = fi & 7;
        float4 f = *(const float4*)(X + (size_t)(m0 + row) * HIDDEN + k0 + c4 * 4);
        v4h h = {(_Float16)f.x, (_Float16)f.y, (_Float16)f.z, (_Float16)f.w};
        *(v4h*)(As + row * A_STR + c4 * 4) = h;
      }
    } else {
      const _Float16* X = (const _Float16*)Xv;
#pragma unroll
      for (int i = 0; i < 2; ++i) {
        int ci = tid + i * 256;            // 512 16B chunks
        int row = ci >> 2, cc = ci & 3;
        *(int4*)(As + row * A_STR + cc * 8) =
            *(const int4*)(X + (size_t)(m0 + row) * HIDDEN + k0 + cc * 8);
      }
    }
    // ---- stage B tile (weights are f32) ----
#pragma unroll
    for (int i = 0; i < 4; ++i) {
      int fi = tid + i * 256;              // 1024 float4 chunks
      int row = fi >> 5, c4 = fi & 31;
      float4 f = *(const float4*)(W + (size_t)(k0 + row) * HIDDEN + n0 + c4 * 4);
      v4h h = {(_Float16)f.x, (_Float16)f.y, (_Float16)f.z, (_Float16)f.w};
      *(v4h*)(Bs + row * B_STR + c4 * 4) = h;
    }
    // ---- prefetch next K-tile into GL2 while this tile computes ----
    if (k0 + BLK_K < HIDDEN) {
      const char* Xb = (const char*)Xv;
      __builtin_prefetch(Xb + ((size_t)(m0 + (tid & 127)) * HIDDEN + k0 + BLK_K) *
                                  (IN_F32 ? 4 : 2), 0, 1);
      __builtin_prefetch(W + (size_t)(k0 + BLK_K + (tid & 31)) * HIDDEN + n0 +
                             ((tid >> 5) & 7) * 16, 0, 1);
    }
    __syncthreads();

    // ---- compute ----
    v16h a[2];
#pragma unroll
    for (int wm = 0; wm < 2; ++wm) {
      int row = waveM * 32 + wm * 16 + (lane & 15);
      a[wm] = load_a_frag(As + row * A_STR, g);
    }
#pragma unroll
    for (int nf = 0; nf < 4; ++nf) {
      v16h b = load_b_frag(Bs + waveN * 64 + nf * 16 + (lane & 15), B_STR, g);
#pragma unroll
      for (int wm = 0; wm < 2; ++wm) acc[wm][nf] = wmma_f16(a[wm], b, acc[wm][nf]);
    }
    __syncthreads();
  }

  // ---- epilogue (C/D layout: lane%16 = N, reg + 8*(lane/16) = M) ----
#pragma unroll
  for (int wm = 0; wm < 2; ++wm) {
#pragma unroll
    for (int nf = 0; nf < 4; ++nf) {
      int n = n0 + waveN * 64 + nf * 16 + (lane & 15);
      float bn = bias[n];
#pragma unroll
      for (int r = 0; r < 8; ++r) {
        int m = m0 + waveM * 32 + wm * 16 + r + g * 8;
        float val = acc[wm][nf][r] + bn;
        if constexpr (OUT_HEADS) {
          val *= scale;  // Q pre-scaled by 1/sqrt(DHEAD)
          int bb = m >> 11, s = m & (SEQ - 1);
          int hh = n >> 6, dd = n & 63;
          ((_Float16*)Ov)[(((size_t)(bb * NHEAD + hh)) * SEQ + s) * DHEAD + dd] =
              (_Float16)val;
        } else {
          ((float*)Ov)[(size_t)m * HIDDEN + n] = val;
        }
      }
    }
  }
}

// ---------------------------------------------------------------------------
// Flash attention, scores^T orientation.
// Block = 128 query rows of one (b,h); 8 waves; wave w owns q columns w*16..+15.
// scores^T: M = key (4 frags), N = q (this wave's 16), K = d (2 wmma steps).
//  - Q^T staged once per block into LDS -> B-fragments are loop-invariant
//  - K/V chunks (64 keys) staged by global_load_async_to_lds_b128, double
//    buffered: issue next chunk, s_wait_asynccnt 4 drains current (in-order)
//  - softmax stats per lane (q = lane%16): in-lane fold + one xor16 shuffle
//  - P stored as [q][key]: 4 packed 16B ds_store per lane; A-frag reload
// ---------------------------------------------------------------------------
#define KCHUNK 64
#define QT_STR 136
#define KS_STR 72
#define VS_STR 72
#define PS_STR 72

__global__ __launch_bounds__(256) void flash_attn(
    const _Float16* __restrict__ Qh, const _Float16* __restrict__ Kh,
    const _Float16* __restrict__ Vh, const int* __restrict__ mask,
    _Float16* __restrict__ ctx) {
  __shared__ _Float16 Qt[DHEAD * QT_STR];          // 64 d x 128 q (transposed)
  __shared__ _Float16 Ks[2][KCHUNK * KS_STR];      // double buffered
  __shared__ _Float16 Vs[2][KCHUNK * VS_STR];
  __shared__ _Float16 Ps[8 * 16 * PS_STR];         // per-wave P scratch

  const int tid  = threadIdx.x;
  const int lane = tid & 31;
  const int wave = tid >> 5;
  const int g    = lane >> 4;
  const int hf   = lane & 15;
  const int q0   = blockIdx.x * 128;
  const int h    = blockIdx.y;
  const int bb   = blockIdx.z;

  const size_t headBase = ((size_t)(bb * NHEAD + h)) * SEQ * DHEAD;

  // ---- stage Q^T once: Qt[d][q] ----
  {
    int q = tid >> 1, dg = tid & 1;
    const _Float16* qp = Qh + headBase + (size_t)(q0 + q) * DHEAD + dg * 32;
#pragma unroll
    for (int j = 0; j < 32; ++j) Qt[(dg * 32 + j) * QT_STR + q] = qp[j];
  }

  // ---- issue async copy of one K/V chunk (wave covers rows w*8..w*8+7) ----
  auto issue_chunk = [&](int kc, int buf) {
    int row0 = wave * 8 + (lane >> 3);
    int col  = (lane & 7) * 8;  // halves
#pragma unroll
    for (int i = 0; i < 2; ++i) {
      int row = row0 + i * 4;
      async_b128((unsigned)(uintptr_t)(&Ks[buf][row * KS_STR + col]),
                 Kh + headBase + (size_t)(kc + row) * DHEAD + col);
      async_b128((unsigned)(uintptr_t)(&Vs[buf][row * VS_STR + col]),
                 Vh + headBase + (size_t)(kc + row) * DHEAD + col);
    }
  };
  issue_chunk(0, 0);
  __syncthreads();  // Qt ready (async K/V covered by per-chunk waits below)

  // loop-invariant Q B-fragments for this wave's 16 q columns
  v16h bq[2];
  bq[0] = load_b_frag(Qt + wave * 16 + hf, QT_STR, g);
  bq[1] = load_b_frag(Qt + 32 * QT_STR + wave * 16 + hf, QT_STR, g);

  const v8f vz = {0.f, 0.f, 0.f, 0.f, 0.f, 0.f, 0.f, 0.f};
  float m_s = -3.0e38f, l_s = 0.f;
  v8f acc_o[4];
#pragma unroll
  for (int nf = 0; nf < 4; ++nf) acc_o[nf] = vz;

  _Float16* PsW = Ps + wave * 16 * PS_STR;
  const int* mrow = mask + (size_t)bb * SEQ * SEQ +
                    (size_t)(q0 + wave * 16 + hf) * SEQ;

  for (int it = 0, kc = 0; kc < SEQ; kc += KCHUNK, ++it) {
    const int cur = it & 1;
    const bool hasNext = (kc + KCHUNK) < SEQ;
    if (hasNext) {
      issue_chunk(kc + KCHUNK, cur ^ 1);
      asm volatile("s_wait_asynccnt 0x4" ::: "memory");  // drain current buf
    } else {
      asm volatile("s_wait_asynccnt 0x0" ::: "memory");
    }
    __syncthreads();

    // ---- scores^T: key rows from Ks (A), q columns from Qt (B) ----
    v8f sc[4];
#pragma unroll
    for (int mf = 0; mf < 4; ++mf) {
      const _Float16* krow = &Ks[cur][(mf * 16 + hf) * KS_STR];
      v16h a0 = load_a_frag(krow, g);
      v16h a1 = load_a_frag(krow + 32, g);
      sc[mf] = wmma_f16(a0, bq[0], vz);
      sc[mf] = wmma_f16(a1, bq[1], sc[mf]);
    }

    // ---- mask (reference: mask==0 -> -10000); keys contiguous per (mf,g) ----
#pragma unroll
    for (int mf = 0; mf < 4; ++mf) {
      int4 ma = *(const int4*)(mrow + kc + mf * 16 + g * 8);
      int4 mb = *(const int4*)(mrow + kc + mf * 16 + g * 8 + 4);
      if (ma.x == 0) sc[mf][0] = -10000.f;
      if (ma.y == 0) sc[mf][1] = -10000.f;
      if (ma.z == 0) sc[mf][2] = -10000.f;
      if (ma.w == 0) sc[mf][3] = -10000.f;
      if (mb.x == 0) sc[mf][4] = -10000.f;
      if (mb.y == 0) sc[mf][5] = -10000.f;
      if (mb.z == 0) sc[mf][6] = -10000.f;
      if (mb.w == 0) sc[mf][7] = -10000.f;
    }

    // ---- online softmax, per lane (q = lane%16), halves merged by xor16 ----
    float cm = -3.0e38f;
#pragma unroll
    for (int mf = 0; mf < 4; ++mf)
#pragma unroll
      for (int r = 0; r < 8; ++r) cm = fmaxf(cm, sc[mf][r]);
    cm = fmaxf(cm, __shfl_xor(cm, 16, 32));
    float mn = fmaxf(m_s, cm);
    float alpha = __expf(m_s - mn);
    m_s = mn;
    float rs = 0.f;
#pragma unroll
    for (int mf = 0; mf < 4; ++mf)
#pragma unroll
      for (int r = 0; r < 8; ++r) {
        sc[mf][r] = __expf(sc[mf][r] - mn);
        rs += sc[mf][r];
      }
    rs += __shfl_xor(rs, 16, 32);
    l_s = l_s * alpha + rs;

    // ---- rescale O (rows are q = r + 8*g: fetch that q's alpha by shuffle) --
#pragma unroll
    for (int r = 0; r < 8; ++r) {
      float ar = __shfl(alpha, r + g * 8, 32);
#pragma unroll
      for (int nf = 0; nf < 4; ++nf) acc_o[nf][r] *= ar;
    }

    // ---- P[q][key]: one packed 16B store per mf (keys mf*16+8g .. +7) ----
#pragma unroll
    for (int mf = 0; mf < 4; ++mf) {
      v8h ph = {(_Float16)sc[mf][0], (_Float16)sc[mf][1], (_Float16)sc[mf][2],
                (_Float16)sc[mf][3], (_Float16)sc[mf][4], (_Float16)sc[mf][5],
                (_Float16)sc[mf][6], (_Float16)sc[mf][7]};
      *(v8h*)(PsW + hf * PS_STR + mf * 16 + g * 8) = ph;
    }

    // ---- O += P @ V (A = P rows q, B = V rows key) ----
#pragma unroll
    for (int kb = 0; kb < 2; ++kb) {
      v16h ap = load_a_frag(PsW + hf * PS_STR + kb * 32, g);
#pragma unroll
      for (int nf = 0; nf < 4; ++nf) {
        v16h bv = load_b_frag(&Vs[cur][(kb * 32) * VS_STR + nf * 16 + hf], VS_STR, g);
        acc_o[nf] = wmma_f16(ap, bv, acc_o[nf]);
      }
    }
    __syncthreads();
  }

  // ---- normalize (l of row q via shuffle) and write ctx [B,S,H] f16 ----
#pragma unroll
  for (int r = 0; r < 8; ++r) {
    float lr = __shfl(l_s, r + g * 8, 32);
    int s = q0 + wave * 16 + r + g * 8;
#pragma unroll
    for (int nf = 0; nf < 4; ++nf) {
      int d = nf * 16 + hf;
      ctx[((size_t)(bb * SEQ + s)) * HIDDEN + h * DHEAD + d] =
          (_Float16)(acc_o[nf][r] / lr);
    }
  }
}

// ---------------------------------------------------------------------------
extern "C" void kernel_launch(void* const* d_in, const int* in_sizes, int n_in,
                              void* d_out, int out_size, void* d_ws, size_t ws_size,
                              hipStream_t stream) {
  const float* q    = (const float*)d_in[0];
  const float* k    = (const float*)d_in[1];
  const float* v    = (const float*)d_in[2];
  const int*   mask = (const int*)d_in[3];
  const float* w_q  = (const float*)d_in[4];
  const float* b_q  = (const float*)d_in[5];
  const float* w_k  = (const float*)d_in[6];
  const float* b_k  = (const float*)d_in[7];
  const float* w_v  = (const float*)d_in[8];
  const float* b_v  = (const float*)d_in[9];
  const float* w_o  = (const float*)d_in[10];
  const float* b_o  = (const float*)d_in[11];

  const size_t headElems = (size_t)BATCH * NHEAD * SEQ * DHEAD;  // 4.19M halves
  _Float16* qh  = (_Float16*)d_ws;
  _Float16* kh  = qh + headElems;
  _Float16* vh  = kh + headElems;
  _Float16* ctx = vh + headElems;   // total ws use: 32 MB

  dim3 blk(256);
  dim3 gg(MTOT / BLK_M, HIDDEN / BLK_N);  // 32 x 8

  gemm_wmma<true, true><<<gg, blk, 0, stream>>>(q, w_q, b_q, qh, 0.125f);
  gemm_wmma<true, true><<<gg, blk, 0, stream>>>(k, w_k, b_k, kh, 1.0f);
  gemm_wmma<true, true><<<gg, blk, 0, stream>>>(v, w_v, b_v, vh, 1.0f);
  flash_attn<<<dim3(SEQ / 128, NHEAD, BATCH), blk, 0, stream>>>(qh, kh, vh, mask, ctx);
  gemm_wmma<false, false><<<gg, blk, 0, stream>>>(ctx, w_o, b_o, d_out, 1.0f);

  (void)in_sizes; (void)n_in; (void)out_size; (void)ws_size;
}
